// Attention_23295902614136
// MI455X (gfx1250) — matrix operand order repaired and checked by reading, not compiled, not run on hardware
//
#include <hip/hip_runtime.h>

typedef __attribute__((ext_vector_type(16))) __bf16 v16bf;
typedef __attribute__((ext_vector_type(8)))  float v8f;
using u16 = unsigned short;
using u32 = unsigned int;

// Problem constants
constexpr int Bb = 2, Nn = 2048, Cc = 1024, Hh = 16, Dd = 64;
constexpr int M_ROWS = Bb * Nn;          // 4096
constexpr int C3 = 3 * Cc;               // 3072
constexpr float ATT_SCALE = 0.125f;      // 64^-0.5

static __device__ __forceinline__ u16 f2bf(float f) {
    union { __bf16 h; u16 u; } c; c.h = (__bf16)f; return c.u;
}

// 16B async copy global -> LDS (CDNA5, tracked by ASYNCcnt)
static __device__ __forceinline__ void async_cp16(u32 lds_off, const void* gaddr) {
    asm volatile("global_load_async_to_lds_b128 %0, %1, off"
                 :: "v"(lds_off), "v"(gaddr) : "memory");
}
static __device__ __forceinline__ void wait_async_le3() {
    asm volatile("s_wait_asynccnt 0x3" ::: "memory");
}
static __device__ __forceinline__ void wait_async_0() {
    asm volatile("s_wait_asynccnt 0x0" ::: "memory");
}

// ---- fragment loaders (wave32 VGPR layouts per CDNA5 ISA 7.12.2) ----------
// A-matrix 16x32 bf16: lane L -> row M = L%16; half = L/16;
//   VGPR v<4: K = half*8 + 2v,2v+1 ; v>=4: K = 16 + half*8 + 2(v-4), +1
static __device__ __forceinline__ v16bf load_a_frag(const u16* base, int row0,
                                                    int ld, int kk, int lane) {
    const int l16 = lane & 15, half = lane >> 4;
    union { v16bf v; u32 u[8]; } f;
    const u32* p = (const u32*)(base + (size_t)(row0 + l16) * ld + kk);
#pragma unroll
    for (int i = 0; i < 8; ++i) {
        int d = (i < 4) ? (half * 8 + 2 * i) : (16 + half * 8 + 2 * (i - 4));
        f.u[i] = p[d >> 1];
    }
    return f.v;
}

// B-matrix 32x16 bf16 where B[k][n] = W[(col0+n)*ld + kk + k] (W row-major):
// lane L -> col N = L%16; half = L/16; VGPR v<4: K = half*8 + 2v,2v+1 ; v>=4: K = 16 + half*8 + 2(v-4), +1
static __device__ __forceinline__ v16bf load_b_frag_rows(const u16* base, int col0,
                                                         int ld, int kk, int lane) {
    const int l16 = lane & 15, half = lane >> 4;
    union { v16bf v; u32 u[8]; } f;
    const u32* p = (const u32*)(base + (size_t)(col0 + l16) * ld + kk);
#pragma unroll
    for (int i = 0; i < 8; ++i) {
        int k = (i < 4) ? (half * 8 + 2 * i) : (16 + half * 8 + 2 * (i - 4));
        f.u[i] = p[k >> 1];
    }
    return f.v;
}

static __device__ __forceinline__ v8f wmma_bf16(v16bf a, v16bf b, v8f c) {
    return __builtin_amdgcn_wmma_f32_16x16x32_bf16(false, a, false, b,
                                                   (short)0, c, false, false);
}

// ---- fp32 -> bf16 conversion --------------------------------------------
__global__ void cvt_bf16_kernel(const float* __restrict__ s, u16* __restrict__ d, int n) {
    int i = blockIdx.x * blockDim.x + threadIdx.x;
    int stride = gridDim.x * blockDim.x;
    for (; i < n; i += stride) d[i] = f2bf(s[i]);
}

// ==========================================================================
// Block-cooperative GEMM skeleton: 256 threads = 8 waves (2x4), block tile
// BM=64 x BN=128, BK=32; async global->LDS double buffering.
// ==========================================================================
constexpr int BM = 64, BN = 128, BK = 32;
constexpr int A_ELEMS = BM * BK;          // 2048 u16
constexpr int B_ELEMS = BN * BK;          // 4096 u16

// ---- QKV GEMM: qkv = xb @ Wqkv^T, scatter to q(scaled)/k [B,H,N,D], vT [B,H,D,N]
__global__ __launch_bounds__(256) void qkv_gemm_kernel(
    const u16* __restrict__ xb, const u16* __restrict__ wq,
    u16* __restrict__ q, u16* __restrict__ k, u16* __restrict__ vT) {
    __shared__ u16 smem[2][A_ELEMS + B_ELEMS];
    const int tid = threadIdx.x;
    const int lane = tid & 31, wid = tid >> 5;
    const int l16 = lane & 15, half = lane >> 4;
    const int wrow = wid >> 2, wcol = wid & 3;

    const int bidn = blockIdx.x % (C3 / BN);     // 24 col blocks
    const int bidm = blockIdx.x / (C3 / BN);     // 64 row blocks
    const int row0 = bidm * BM, col0 = bidn * BN;

    auto issue = [&](int buf, int kk) {
        u16* sA = smem[buf];
        u16* sB = smem[buf] + A_ELEMS;
        {   // A: 64 rows x 4 x 16B chunks = 256 chunks, 1/thread
            const int row = tid >> 2, seg = tid & 3;
            async_cp16((u32)(uintptr_t)(sA + row * BK + seg * 8),
                       xb + (size_t)(row0 + row) * Cc + kk + seg * 8);
        }
#pragma unroll
        for (int c = 0; c < 2; ++c) {   // B: 128 rows x 4 chunks = 512, 2/thread
            const int idx = tid + c * 256;
            const int row = idx >> 2, seg = idx & 3;
            async_cp16((u32)(uintptr_t)(sB + row * BK + seg * 8),
                       wq + (size_t)(col0 + row) * Cc + kk + seg * 8);
        }
    };

    v8f acc[2][2] = {};
    issue(0, 0);
    for (int it = 0; it < Cc / BK; ++it) {       // 32 K-steps
        const int cur = it & 1;
        if (it + 1 < Cc / BK) { issue(cur ^ 1, (it + 1) * BK); wait_async_le3(); }
        else                  { wait_async_0(); }
        __syncthreads();
        const u16* sA = smem[cur];
        const u16* sB = smem[cur] + A_ELEMS;
        v16bf a0 = load_a_frag(sA, wrow * 32,      BK, 0, lane);
        v16bf a1 = load_a_frag(sA, wrow * 32 + 16, BK, 0, lane);
        v16bf b0 = load_b_frag_rows(sB, wcol * 32,      BK, 0, lane);
        v16bf b1 = load_b_frag_rows(sB, wcol * 32 + 16, BK, 0, lane);
        acc[0][0] = wmma_bf16(a0, b0, acc[0][0]);
        acc[0][1] = wmma_bf16(a0, b1, acc[0][1]);
        acc[1][0] = wmma_bf16(a1, b0, acc[1][0]);
        acc[1][1] = wmma_bf16(a1, b1, acc[1][1]);
        __syncthreads();
    }

    // epilogue: route each 16-col group to q (scaled) / k / vT
#pragma unroll
    for (int s = 0; s < 2; ++s) {
#pragma unroll
        for (int t = 0; t < 2; ++t) {
            const int oc0 = col0 + wcol * 32 + t * 16;   // 16-aligned, one (tsel,h)
            const int tsel = oc0 >> 10;
            const int h = (oc0 >> 6) & 15;
            const int d = (oc0 & 63) + l16;
#pragma unroll
            for (int r = 0; r < 8; ++r) {
                const int mrow = row0 + wrow * 32 + s * 16 + half * 8 + r;
                const int b = mrow >> 11;
                const int n = mrow & (Nn - 1);
                const float val = acc[s][t][r];
                const size_t hb = (size_t)(b * Hh + h);
                if (tsel == 0)      q[(hb * Nn + n) * Dd + d] = f2bf(val * ATT_SCALE);
                else if (tsel == 1) k[(hb * Nn + n) * Dd + d] = f2bf(val);
                else                vT[(hb * Dd + d) * Nn + n] = f2bf(val);
            }
        }
    }
}

// ---- Flash attention: one wave per (b, h, 16-query tile) -----------------
__global__ __launch_bounds__(256) void attn_kernel(
    const u16* __restrict__ q, const u16* __restrict__ k,
    const u16* __restrict__ vT, u16* __restrict__ ao) {
    __shared__ u16 pshm[8][16 * 32];             // per-wave P-tile transpose buffer
    const int lane = threadIdx.x & 31;
    const int wid = threadIdx.x >> 5;
    const int l16 = lane & 15, half = lane >> 4;
    const int w = blockIdx.x * 8 + wid;          // 4096 waves total
    const int qt = w & 127;
    const int h = (w >> 7) & 15;
    const int b = w >> 11;
    const int qrow0 = qt * 16;

    const u16* qh = q  + ((size_t)b * Hh + h) * Nn * Dd;
    const u16* kh = k  + ((size_t)b * Hh + h) * Nn * Dd;
    const u16* vh = vT + ((size_t)b * Hh + h) * Dd * Nn;   // [D][N]

    v16bf qa0 = load_a_frag(qh, qrow0, Dd, 0, lane);
    v16bf qa1 = load_a_frag(qh, qrow0, Dd, 32, lane);

    float m_run[8], l_run[8];
#pragma unroll
    for (int r = 0; r < 8; ++r) { m_run[r] = -1e30f; l_run[r] = 0.0f; }
    v8f o[4] = {};

    for (int jb = 0; jb < Nn; jb += 32) {
        if (jb + 32 < Nn) {
            __builtin_prefetch(kh + (size_t)(jb + 32 + l16) * Dd, 0, 1);
            __builtin_prefetch(vh + (size_t)(l16)*Nn + jb + 32, 0, 1);
        }
        // ---- issue ALL global fragment loads up front (K and V) ----
        v16bf kb00 = load_b_frag_rows(kh, jb,      Dd, 0,  lane);
        v16bf kb01 = load_b_frag_rows(kh, jb,      Dd, 32, lane);
        v16bf kb10 = load_b_frag_rows(kh, jb + 16, Dd, 0,  lane);
        v16bf kb11 = load_b_frag_rows(kh, jb + 16, Dd, 32, lane);
        v16bf vb0  = load_b_frag_rows(vh, 0,  Nn, jb, lane);   // d cols 0..15
        v16bf vb1  = load_b_frag_rows(vh, 16, Nn, jb, lane);
        v16bf vb2  = load_b_frag_rows(vh, 32, Nn, jb, lane);
        v16bf vb3  = load_b_frag_rows(vh, 48, Nn, jb, lane);

        // scores: two 16x16 tiles, K-dim 64 = 2 chained WMMAs each
        v8f s0 = {}, s1 = {};
        s0 = wmma_bf16(qa0, kb00, s0);
        s0 = wmma_bf16(qa1, kb01, s0);
        s1 = wmma_bf16(qa0, kb10, s1);
        s1 = wmma_bf16(qa1, kb11, s1);

        // online softmax; C-layout: lane holds rows half*8+r, col = l16 (+16)
        float corr[8];
#pragma unroll
        for (int r = 0; r < 8; ++r) {
            float mx = fmaxf(s0[r], s1[r]);
#pragma unroll
            for (int off = 8; off >= 1; off >>= 1)
                mx = fmaxf(mx, __shfl_xor(mx, off, 32));
            const float mn = fmaxf(m_run[r], mx);
            corr[r] = __expf(m_run[r] - mn);
            const float p0 = __expf(s0[r] - mn);
            const float p1 = __expf(s1[r] - mn);
            s0[r] = p0; s1[r] = p1;
            float rs = p0 + p1;
#pragma unroll
            for (int off = 8; off >= 1; off >>= 1)
                rs += __shfl_xor(rs, off, 32);
            l_run[r] = l_run[r] * corr[r] + rs;
            m_run[r] = mn;
        }
#pragma unroll
        for (int g = 0; g < 4; ++g)
#pragma unroll
            for (int r = 0; r < 8; ++r) o[g][r] *= corr[r];

        // transpose P (C-layout -> A-layout) through per-wave LDS (in-order)
        u16* pb = pshm[wid];
#pragma unroll
        for (int r = 0; r < 8; ++r) {
            const int row = half * 8 + r;
            pb[row * 32 + l16]      = f2bf(s0[r]);
            pb[row * 32 + 16 + l16] = f2bf(s1[r]);
        }
        v16bf pa = load_a_frag(pb, 0, 32, 0, lane);

        o[0] = wmma_bf16(pa, vb0, o[0]);
        o[1] = wmma_bf16(pa, vb1, o[1]);
        o[2] = wmma_bf16(pa, vb2, o[2]);
        o[3] = wmma_bf16(pa, vb3, o[3]);
    }

    // epilogue: O /= l ; write [B,N,C] bf16 head-major (h*64+d)
#pragma unroll
    for (int r = 0; r < 8; ++r) {
        const float inv = 1.0f / l_run[r];
        const int qr = qrow0 + half * 8 + r;
#pragma unroll
        for (int g = 0; g < 4; ++g) {
            const int d = g * 16 + l16;
            ao[((size_t)b * Nn + qr) * Cc + h * Dd + d] = f2bf(o[g][r] * inv);
        }
    }
}

// ---- Proj GEMM (block-cooperative, async double-buffered): out = ao@Wp^T + b
__global__ __launch_bounds__(256) void proj_gemm_kernel(
    const u16* __restrict__ ao, const u16* __restrict__ wp,
    const float* __restrict__ bias, float* __restrict__ out) {
    __shared__ u16 smem[2][A_ELEMS + B_ELEMS];
    const int tid = threadIdx.x;
    const int lane = tid & 31, wid = tid >> 5;
    const int l16 = lane & 15, half = lane >> 4;
    const int wrow = wid >> 2, wcol = wid & 3;

    const int bidn = blockIdx.x % (Cc / BN);     // 8 col blocks
    const int bidm = blockIdx.x / (Cc / BN);     // 64 row blocks
    const int row0 = bidm * BM, col0 = bidn * BN;

    auto issue = [&](int buf, int kk) {
        u16* sA = smem[buf];
        u16* sB = smem[buf] + A_ELEMS;
        {
            const int row = tid >> 2, seg = tid & 3;
            async_cp16((u32)(uintptr_t)(sA + row * BK + seg * 8),
                       ao + (size_t)(row0 + row) * Cc + kk + seg * 8);
        }
#pragma unroll
        for (int c = 0; c < 2; ++c) {
            const int idx = tid + c * 256;
            const int row = idx >> 2, seg = idx & 3;
            async_cp16((u32)(uintptr_t)(sB + row * BK + seg * 8),
                       wp + (size_t)(col0 + row) * Cc + kk + seg * 8);
        }
    };

    v8f acc[2][2] = {};
    issue(0, 0);
    for (int it = 0; it < Cc / BK; ++it) {
        const int cur = it & 1;
        if (it + 1 < Cc / BK) { issue(cur ^ 1, (it + 1) * BK); wait_async_le3(); }
        else                  { wait_async_0(); }
        __syncthreads();
        const u16* sA = smem[cur];
        const u16* sB = smem[cur] + A_ELEMS;
        v16bf a0 = load_a_frag(sA, wrow * 32,      BK, 0, lane);
        v16bf a1 = load_a_frag(sA, wrow * 32 + 16, BK, 0, lane);
        v16bf b0 = load_b_frag_rows(sB, wcol * 32,      BK, 0, lane);
        v16bf b1 = load_b_frag_rows(sB, wcol * 32 + 16, BK, 0, lane);
        acc[0][0] = wmma_bf16(a0, b0, acc[0][0]);
        acc[0][1] = wmma_bf16(a0, b1, acc[0][1]);
        acc[1][0] = wmma_bf16(a1, b0, acc[1][0]);
        acc[1][1] = wmma_bf16(a1, b1, acc[1][1]);
        __syncthreads();
    }

#pragma unroll
    for (int s = 0; s < 2; ++s) {
#pragma unroll
        for (int t = 0; t < 2; ++t) {
            const int oc = col0 + wcol * 32 + t * 16 + l16;
            const float bi = bias[oc];
#pragma unroll
            for (int r = 0; r < 8; ++r) {
                const int mrow = row0 + wrow * 32 + s * 16 + half * 8 + r;
                out[(size_t)mrow * Cc + oc] = acc[s][t][r] + bi;
            }
        }
    }
}

extern "C" void kernel_launch(void* const* d_in, const int* in_sizes, int n_in,
                              void* d_out, int out_size, void* d_ws, size_t ws_size,
                              hipStream_t stream) {
    const float* x     = (const float*)d_in[0];
    const float* Wqkv  = (const float*)d_in[1];
    const float* Wproj = (const float*)d_in[2];
    const float* bproj = (const float*)d_in[3];
    float* out = (float*)d_out;

    // workspace carve-up (bf16 buffers)
    char* p = (char*)d_ws;
    u16* xb    = (u16*)p;               p += (size_t)M_ROWS * Cc * 2;   // 8 MB
    u16* wqkvb = (u16*)p;               p += (size_t)C3 * Cc * 2;       // 6 MB
    u16* wprojb= (u16*)p;               p += (size_t)Cc * Cc * 2;       // 2 MB
    u16* q     = (u16*)p;               p += (size_t)Bb * Hh * Nn * Dd * 2;
    u16* k     = (u16*)p;               p += (size_t)Bb * Hh * Nn * Dd * 2;
    u16* vT    = (u16*)p;               p += (size_t)Bb * Hh * Nn * Dd * 2;
    u16* ao    = (u16*)p;

    cvt_bf16_kernel<<<1024, 256, 0, stream>>>(x, xb, M_ROWS * Cc);
    cvt_bf16_kernel<<<1024, 256, 0, stream>>>(Wqkv, wqkvb, C3 * Cc);
    cvt_bf16_kernel<<<1024, 256, 0, stream>>>(Wproj, wprojb, Cc * Cc);

    // (4096/64) row blocks * (3072/128) col blocks = 1536
    qkv_gemm_kernel<<<1536, 256, 0, stream>>>(xb, wqkvb, q, k, vT);
    // B*H*(N/16) = 4096 waves / 8 per block
    attn_kernel<<<512, 256, 0, stream>>>(q, k, vT, ao);
    // (4096/64) * (1024/128) = 512
    proj_gemm_kernel<<<512, 256, 0, stream>>>(ao, wprojb, bproj, out);
}
